// SlotAttention_36447092474196
// MI455X (gfx1250) — compile-verified, matrix operand order checked by reading, not run on hardware
//
#include <hip/hip_runtime.h>

// ---------------------------------------------------------------------------
// Slot Attention for MI455X (gfx1250, wave32, WMMA bf16 16x16x32).
// k/v stored bf16 (134MB -> L2-resident across the 3 iterations), fp32
// accumulate via v_wmma_f32_16x16x32_bf16, async-to/from-LDS (ASYNCcnt) for
// tile staging, fp32 VALU for the tiny GRU/MLP tail.
// ---------------------------------------------------------------------------

#define TB 64
#define TN 4096
#define TH 128
#define TS 7
#define TM 256
#define N_ITERS 3
#define LN_EPS 1e-5f
#define ATT_EPS 1e-8f

typedef unsigned int u32;
typedef unsigned long long u64;
typedef unsigned short u16;
typedef __bf16 bf16;
typedef bf16 v16bf __attribute__((ext_vector_type(16)));
typedef float v8f __attribute__((ext_vector_type(8)));

union AFrag { v16bf v; u32 d[8]; };

__device__ __forceinline__ u16 f2bf(float f) {
  union { float f; u32 u; } v; v.f = f;
  u32 r = v.u + 0x7FFFu + ((v.u >> 16) & 1u);   // round-to-nearest-even
  return (u16)(r >> 16);
}

// 16-bit WMMA operand packing (ISA 7.12.2): lane half = lane>>4.
// dword p of the v16bf holds the K pair starting at 2*dwIdx of a 32-K step.
__device__ __forceinline__ int dwIdx(int p, int half) {
  return p + (p >= 4 ? 4 : 0) + half * 4;
}

// Low 32 bits of the flat shared address are the LDS byte offset (ISA 10.2:
// aperture is addr[63:32]).
__device__ __forceinline__ u32 lds_off(const void* p) {
  return (u32)(uintptr_t)p;
}

__device__ __forceinline__ void async_load_b128(u32 lds_addr, u64 gaddr) {
  asm volatile("global_load_async_to_lds_b128 %0, %1, off"
               :: "v"(lds_addr), "v"(gaddr) : "memory");
}
__device__ __forceinline__ void async_store_b128(u64 gaddr, u32 lds_addr) {
  asm volatile("global_store_async_from_lds_b128 %0, %1, off"
               :: "v"(gaddr), "v"(lds_addr) : "memory");
}

// ---------------------------------------------------------------------------
// Kernel 0: fp32 -> bf16 weight conversion (Wk, Wv), stays hot in L2.
// ---------------------------------------------------------------------------
__global__ __launch_bounds__(256) void k_cvt_w(const float* __restrict__ Wk,
                                               const float* __restrict__ Wv,
                                               u16* __restrict__ wkbf,
                                               u16* __restrict__ wvbf) {
  int i = blockIdx.x * 256 + threadIdx.x;
  if (i < TH * TH) { wkbf[i] = f2bf(Wk[i]); wvbf[i] = f2bf(Wv[i]); }
}

// ---------------------------------------------------------------------------
// Kernel 1: fused input LayerNorm + k = x@Wk^T, v = x@Wv^T  (bf16 out).
// 256 threads (8 waves), 64 rows. Wave w: matrix (w&1), row-tile (w>>1).
// D tiles staged in LDS, drained with global_store_async_from_lds_b128.
// ---------------------------------------------------------------------------
__global__ __launch_bounds__(256) void k_ln_project(
    const float* __restrict__ x, const float* __restrict__ g,
    const float* __restrict__ bia, const u16* __restrict__ wkbf,
    const u16* __restrict__ wvbf, u16* __restrict__ kout,
    u16* __restrict__ vout) {
  __shared__ u32 xln[64 * 65];        // 64 rows x 64 bf16 pairs, stride 65 dw
  __shared__ u32 outstage[2][64 * 64];  // k|v staging, 64 rows x 256B, packed
  __shared__ float rsum[64][4];
  __shared__ float rsq[64][4];

  const int tid = threadIdx.x;
  const long long row0 = (long long)blockIdx.x * 64;

  // ---- LayerNorm: 4 threads per row, 32 elements each ----
  {
    int row = tid >> 2, qr = tid & 3;
    const float* xr = x + (row0 + row) * TH + qr * 32;
    float v[32];
#pragma unroll
    for (int i = 0; i < 8; ++i) {
      float4 t = ((const float4*)xr)[i];
      v[i * 4 + 0] = t.x; v[i * 4 + 1] = t.y;
      v[i * 4 + 2] = t.z; v[i * 4 + 3] = t.w;
    }
    float s = 0.f, s2 = 0.f;
#pragma unroll
    for (int i = 0; i < 32; ++i) { s += v[i]; s2 += v[i] * v[i]; }
    rsum[row][qr] = s; rsq[row][qr] = s2;
    __syncthreads();
    float mean = (rsum[row][0] + rsum[row][1] + rsum[row][2] + rsum[row][3]) *
                 (1.f / TH);
    float var = (rsq[row][0] + rsq[row][1] + rsq[row][2] + rsq[row][3]) *
                    (1.f / TH) - mean * mean;
    float rstd = rsqrtf(var + LN_EPS);
#pragma unroll
    for (int i = 0; i < 16; ++i) {
      int d0 = qr * 32 + i * 2;
      float a = (v[i * 2] - mean) * rstd * g[d0] + bia[d0];
      float b = (v[i * 2 + 1] - mean) * rstd * g[d0 + 1] + bia[d0 + 1];
      xln[row * 65 + (d0 >> 1)] = (u32)f2bf(a) | ((u32)f2bf(b) << 16);
    }
  }
  __syncthreads();

  // ---- WMMA: each wave computes one 16x128 tile of k or v ----
  const int wave = tid >> 5, lane = tid & 31;
  const int half = lane >> 4, l16 = lane & 15;
  const int matsel = wave & 1, rt = wave >> 1;
  const u32* w32 = (const u32*)(matsel ? wvbf : wkbf);
  u16* osel = matsel ? vout : kout;
  u16* os16 = (u16*)&outstage[matsel][0];

  AFrag A[4];
#pragma unroll
  for (int ks = 0; ks < 4; ++ks)
#pragma unroll
    for (int p = 0; p < 8; ++p)
      A[ks].d[p] = xln[(rt * 16 + l16) * 65 + ks * 16 + dwIdx(p, half)];

  for (int ct = 0; ct < 8; ++ct) {
    v8f acc = {};
#pragma unroll
    for (int ks = 0; ks < 4; ++ks) {
      AFrag Bf;
#pragma unroll
      for (int p = 0; p < 8; ++p)
        Bf.d[p] = w32[(ct * 16 + l16) * 64 + ks * 16 + dwIdx(p, half)];
      acc = __builtin_amdgcn_wmma_f32_16x16x32_bf16(
          false, A[ks].v, false, Bf.v, (short)0, acc, false, false);
    }
#pragma unroll
    for (int r = 0; r < 8; ++r)   // D: M = r + half*8, N = lane&15
      os16[(rt * 16 + r + half * 8) * 128 + ct * 16 + l16] = f2bf(acc[r]);
  }

  // Drain this wave's 16 rows (4KB) with async B128 stores from LDS.
  asm volatile("s_wait_dscnt 0" ::: "memory");   // wave's DS writes visible
  const u32 os_base = lds_off(os16);
#pragma unroll
  for (int j = 0; j < 8; ++j) {
    int c = lane * 8 + j;               // 256 chunks: 16 rows x 16 segs
    int rl = c >> 4, seg = c & 15;
    u64 ga = (u64)(uintptr_t)osel + ((u64)(row0 + rt * 16 + rl) << 8) +
             (u64)(seg << 4);
    async_store_b128(ga, os_base + (rt * 16 + rl) * 256 + seg * 16);
  }
  // s_endpgm performs an implicit wait-idle (covers ASYNCcnt).
}

// ---------------------------------------------------------------------------
// Kernel 2: slots = mu + exp(log_sigma) * slots_init   (first 128 of 132 ch)
// ---------------------------------------------------------------------------
__global__ __launch_bounds__(128) void k_init_slots(
    const float* __restrict__ init, const float* __restrict__ mu,
    const float* __restrict__ ls, float* __restrict__ slots) {
  int b = blockIdx.x, h = threadIdx.x;
  float m = mu[h], e = __expf(ls[h]);
  for (int s = 0; s < TS; ++s)
    slots[(b * TS + s) * TH + h] = m + e * init[(b * TS + s) * (TH + 4) + h];
}

// ---------------------------------------------------------------------------
// Kernel 3 (per iter): slot LayerNorm + q = sn@Wq^T (scale folded), bf16 out
// padded to 16 slot rows (zeros); also zeroes U / colsum accumulators.
// ---------------------------------------------------------------------------
__global__ __launch_bounds__(128) void k_slots_pre(
    const float* __restrict__ slots, const float* __restrict__ g,
    const float* __restrict__ bia, const float* __restrict__ Wq,
    u16* __restrict__ qbuf, float* __restrict__ Ubuf,
    float* __restrict__ csum) {
  __shared__ float sn[TS][TH];
  __shared__ float mv[TS][2];
  int b = blockIdx.x, t = threadIdx.x;
  if (t < TS) {
    float s = 0.f, s2 = 0.f;
    for (int d = 0; d < TH; ++d) {
      float v = slots[(b * TS + t) * TH + d];
      s += v; s2 += v * v;
    }
    float mean = s * (1.f / TH);
    mv[t][0] = mean;
    mv[t][1] = rsqrtf(s2 * (1.f / TH) - mean * mean + LN_EPS);
  }
  __syncthreads();
  for (int s = 0; s < TS; ++s)
    sn[s][t] = (slots[(b * TS + s) * TH + t] - mv[s][0]) * mv[s][1] * g[t] +
               bia[t];
  __syncthreads();
  const float scale = 0.08838834764831845f;  // 128^-0.5 folded into q
  for (int s = 0; s < TS; ++s) {
    float acc = 0.f;
    for (int d = 0; d < TH; ++d) acc += sn[s][d] * Wq[t * TH + d];
    qbuf[(b * 16 + s) * TH + t] = f2bf(acc * scale);
  }
  for (int s = TS; s < 16; ++s) qbuf[(b * 16 + s) * TH + t] = 0;
  for (int i = t; i < 16 * TH; i += 128) Ubuf[b * 16 * TH + i] = 0.f;
  if (t < 16) csum[b * 16 + t] = 0.f;
}

// ---------------------------------------------------------------------------
// Kernel 4 (per iter): per 128-row chunk of n:
//   q+v tiles -> LDS via global_load_async_to_lds_b128 (v copy overlaps the
//   logits phase; s_wait_asynccnt 16 releases q, 0 releases v)
//   logits = k @ q^T (WMMA, K=128) -> softmax(7)+eps -> bf16 attn^T in LDS
//   U += attn^T @ v (WMMA, K=32/wave) + colsum, global f32 atomics.
// Block = 128 threads (4 waves), wave owns 32 n-rows.
// ---------------------------------------------------------------------------
__global__ __launch_bounds__(128) void k_attention(
    const u16* __restrict__ kbuf, const u16* __restrict__ vbuf,
    const u16* __restrict__ qbuf, float* __restrict__ Ubuf,
    float* __restrict__ csum) {
  __shared__ u32 qt[16 * 68];        // q tile, 272B row stride (16B aligned)
  __shared__ u32 vt[128 * 68];       // v tile, 272B row stride
  __shared__ float lgs[4][16 * 17];  // per-wave logits tile
  __shared__ u32 attnA[4][16 * 17];  // per-wave attn^T bf16, A-operand layout

  const int b = blockIdx.x >> 5;     // 32 chunks of 128 rows per batch
  const int chunk = blockIdx.x & 31;
  const int tid = threadIdx.x, wave = tid >> 5, lane = tid & 31;
  const int half = lane >> 4, l16 = lane & 15;
  const int nbase = chunk * 128;
  const u32* kb32 = (const u32*)kbuf;

  // Async stage q (2 chunks/thread) then v (16 chunks/thread): in-order
  // completion lets s_wait_asynccnt 16 release q while v is still in flight.
  {
    const u64 qg = (u64)(uintptr_t)(qbuf + (size_t)b * 16 * TH);
    const u32 ql = lds_off(qt);
#pragma unroll
    for (int j = 0; j < 2; ++j) {
      int c = tid + j * 128, row = c >> 4, seg = c & 15;
      async_load_b128(ql + row * 272 + seg * 16, qg + row * 256 + seg * 16);
    }
    const u64 vg = (u64)(uintptr_t)(vbuf + (size_t)(b * TN + nbase) * TH);
    const u32 vl = lds_off(vt);
#pragma unroll
    for (int j = 0; j < 16; ++j) {
      int c = tid + j * 128, row = c >> 4, seg = c & 15;
      async_load_b128(vl + row * 272 + seg * 16, vg + row * 256 + seg * 16);
    }
  }
  for (int i = lane; i < 16 * 17; i += 32) attnA[wave][i] = 0;  // pad s>=7
  asm volatile("s_wait_asynccnt 16" ::: "memory");  // q tile resident
  __syncthreads();

  float cs[TS];
#pragma unroll
  for (int s = 0; s < TS; ++s) cs[s] = 0.f;

  for (int sub = 0; sub < 2; ++sub) {
    const int nrow0 = nbase + wave * 32 + sub * 16;
    v8f acc = {};
#pragma unroll
    for (int ks = 0; ks < 4; ++ks) {
      AFrag Af, Bf;
#pragma unroll
      for (int p = 0; p < 8; ++p) {
        int dw = dwIdx(p, half);
        Af.d[p] = kb32[(size_t)(b * TN + nrow0 + l16) * 64 + ks * 16 + dw];
        Bf.d[p] = qt[l16 * 68 + ks * 16 + dw];
      }
      acc = __builtin_amdgcn_wmma_f32_16x16x32_bf16(
          false, Af.v, false, Bf.v, (short)0, acc, false, false);
    }
#pragma unroll
    for (int r = 0; r < 8; ++r)
      lgs[wave][(r + half * 8) * 17 + l16] = acc[r];
    asm volatile("s_wait_dscnt 0" ::: "memory");  // same-wave DS in-order

    if (lane < 16) {  // one lane per n-row: 7-way softmax
      float lg[TS], mx = -1e30f;
#pragma unroll
      for (int s = 0; s < TS; ++s) {
        lg[s] = lgs[wave][lane * 17 + s];
        mx = fmaxf(mx, lg[s]);
      }
      float sum = 0.f;
#pragma unroll
      for (int s = 0; s < TS; ++s) { lg[s] = __expf(lg[s] - mx); sum += lg[s]; }
      float inv = 1.f / sum;
      u16* aa = (u16*)&attnA[wave][0];
#pragma unroll
      for (int s = 0; s < TS; ++s) {
        float a = lg[s] * inv + ATT_EPS;
        cs[s] += a;
        aa[s * 34 + sub * 16 + lane] = f2bf(a);  // attn^T row s, k = n_local
      }
    }
    asm volatile("s_wait_dscnt 0" ::: "memory");
  }

  asm volatile("s_wait_asynccnt 0" ::: "memory");  // v tile resident
  __syncthreads();

  // U[s,h] += attn^T (16x32) @ v (32x128), one K=32 WMMA per h-tile
  AFrag Aa;
#pragma unroll
  for (int p = 0; p < 8; ++p)
    Aa.d[p] = attnA[wave][l16 * 17 + dwIdx(p, half)];

  const u16* vt16 = (const u16*)vt;
  for (int ct = 0; ct < 8; ++ct) {
    AFrag Bv;
#pragma unroll
    for (int e = 0; e < 16; ++e) {
      int k = e + half * 8 + (e >= 8 ? 8 : 0);  // B: K along elements
      ((u16*)Bv.d)[e] = vt16[(wave * 32 + k) * 136 + ct * 16 + l16];
    }
    v8f acc = {};
    acc = __builtin_amdgcn_wmma_f32_16x16x32_bf16(
        false, Aa.v, false, Bv.v, (short)0, acc, false, false);
#pragma unroll
    for (int r = 0; r < 8; ++r) {
      int s = r + half * 8;
      if (s < TS)
        atomicAdd(&Ubuf[(b * 16 + s) * TH + ct * 16 + l16], acc[r]);
    }
  }
  if (lane < 16) {
#pragma unroll
    for (int s = 0; s < TS; ++s) atomicAdd(&csum[b * 16 + s], cs[s]);
  }
}

// ---------------------------------------------------------------------------
// Kernel 5 (per iter): updates = U/colsum; GRU cell; residual MLP. fp32 VALU.
// ---------------------------------------------------------------------------
__global__ __launch_bounds__(256) void k_gru_mlp(
    const float* __restrict__ Ubuf, const float* __restrict__ csum,
    const float* __restrict__ W_ih, const float* __restrict__ W_hh,
    const float* __restrict__ b_ih, const float* __restrict__ b_hh,
    const float* __restrict__ ln_m_g, const float* __restrict__ ln_m_b,
    const float* __restrict__ w1, const float* __restrict__ b1,
    const float* __restrict__ w2, const float* __restrict__ b2,
    float* __restrict__ slots) {
  __shared__ float upd[TS][TH];
  __shared__ float hprev[TS][TH];
  __shared__ float gi[TS][3 * TH];
  __shared__ float gh[TS][3 * TH];
  __shared__ float snew[TS][TH];
  __shared__ float mn[TS][TH];
  __shared__ float hid[TS][TM];
  __shared__ float mv[TS][2];

  const int b = blockIdx.x, t = threadIdx.x;
  for (int i = t; i < TS * TH; i += 256) {
    int s = i / TH, h = i % TH;
    upd[s][h] = Ubuf[(b * 16 + s) * TH + h] / csum[b * 16 + s];
    hprev[s][h] = slots[(b * TS + s) * TH + h];
  }
  __syncthreads();
  for (int i = t; i < TS * 3 * TH; i += 256) {
    int s = i / (3 * TH), gidx = i % (3 * TH);
    float ai = b_ih[gidx], ah = b_hh[gidx];
    const float* wi = W_ih + gidx * TH;
    const float* wh = W_hh + gidx * TH;
    for (int d = 0; d < TH; ++d) {
      ai += upd[s][d] * wi[d];
      ah += hprev[s][d] * wh[d];
    }
    gi[s][gidx] = ai; gh[s][gidx] = ah;
  }
  __syncthreads();
  for (int i = t; i < TS * TH; i += 256) {
    int s = i / TH, h = i % TH;
    float r = 1.f / (1.f + __expf(-(gi[s][h] + gh[s][h])));
    float z = 1.f / (1.f + __expf(-(gi[s][TH + h] + gh[s][TH + h])));
    float n = tanhf(gi[s][2 * TH + h] + r * gh[s][2 * TH + h]);
    snew[s][h] = (1.f - z) * n + z * hprev[s][h];
  }
  __syncthreads();
  if (t < TS) {
    float s_ = 0.f, s2 = 0.f;
    for (int d = 0; d < TH; ++d) {
      float v = snew[t][d]; s_ += v; s2 += v * v;
    }
    float mean = s_ * (1.f / TH);
    mv[t][0] = mean;
    mv[t][1] = rsqrtf(s2 * (1.f / TH) - mean * mean + LN_EPS);
  }
  __syncthreads();
  for (int i = t; i < TS * TH; i += 256) {
    int s = i / TH, h = i % TH;
    mn[s][h] = (snew[s][h] - mv[s][0]) * mv[s][1] * ln_m_g[h] + ln_m_b[h];
  }
  __syncthreads();
  for (int i = t; i < TS * TM; i += 256) {
    int s = i / TM, m = i % TM;
    float a = b1[m];
    const float* w = w1 + m * TH;
    for (int d = 0; d < TH; ++d) a += mn[s][d] * w[d];
    hid[s][m] = fmaxf(a, 0.f);
  }
  __syncthreads();
  for (int i = t; i < TS * TH; i += 256) {
    int s = i / TH, h = i % TH;
    float a = b2[h];
    const float* w = w2 + h * TM;
    for (int m = 0; m < TM; ++m) a += hid[s][m] * w[m];
    slots[(b * TS + s) * TH + h] = snew[s][h] + a;
  }
}

// ---------------------------------------------------------------------------
extern "C" void kernel_launch(void* const* d_in, const int* in_sizes, int n_in,
                              void* d_out, int out_size, void* d_ws,
                              size_t ws_size, hipStream_t stream) {
  (void)in_sizes; (void)n_in; (void)out_size; (void)ws_size;
  const float* inputs     = (const float*)d_in[0];
  const float* slots_init = (const float*)d_in[1];
  const float* ln_in_g    = (const float*)d_in[2];
  const float* ln_in_b    = (const float*)d_in[3];
  const float* ln_s_g     = (const float*)d_in[4];
  const float* ln_s_b     = (const float*)d_in[5];
  const float* ln_m_g     = (const float*)d_in[6];
  const float* ln_m_b     = (const float*)d_in[7];
  const float* Wq         = (const float*)d_in[8];
  const float* Wk         = (const float*)d_in[9];
  const float* Wv         = (const float*)d_in[10];
  const float* W_ih       = (const float*)d_in[11];
  const float* W_hh       = (const float*)d_in[12];
  const float* b_ih       = (const float*)d_in[13];
  const float* b_hh       = (const float*)d_in[14];
  const float* w1         = (const float*)d_in[15];
  const float* b1         = (const float*)d_in[16];
  const float* w2         = (const float*)d_in[17];
  const float* b2         = (const float*)d_in[18];
  const float* mu         = (const float*)d_in[19];
  const float* ls         = (const float*)d_in[20];
  float* slots = (float*)d_out;

  u16* kbuf = (u16*)d_ws;                       // 64MB bf16
  u16* vbuf = kbuf + (size_t)TB * TN * TH;      // 64MB bf16
  u16* qbuf = vbuf + (size_t)TB * TN * TH;      // 64x16x128 bf16
  u16* wkbf = qbuf + (size_t)TB * 16 * TH;
  u16* wvbf = wkbf + TH * TH;
  float* Ubuf = (float*)(wvbf + TH * TH);       // 64x16x128 f32
  float* csum = Ubuf + (size_t)TB * 16 * TH;    // 64x16 f32

  k_cvt_w<<<64, 256, 0, stream>>>(Wk, Wv, wkbf, wvbf);
  k_ln_project<<<(TB * TN) / 64, 256, 0, stream>>>(inputs, ln_in_g, ln_in_b,
                                                   wkbf, wvbf, kbuf, vbuf);
  k_init_slots<<<TB, 128, 0, stream>>>(slots_init, mu, ls, slots);
  for (int it = 0; it < N_ITERS; ++it) {
    k_slots_pre<<<TB, 128, 0, stream>>>(slots, ln_s_g, ln_s_b, Wq, qbuf, Ubuf,
                                        csum);
    k_attention<<<TB * 32, 128, 0, stream>>>(kbuf, vbuf, qbuf, Ubuf, csum);
    k_gru_mlp<<<TB, 256, 0, stream>>>(Ubuf, csum, W_ih, W_hh, b_ih, b_hh,
                                      ln_m_g, ln_m_b, w1, b1, w2, b2, slots);
  }
}
